// ContentEncoder_12111807774930
// MI455X (gfx1250) — matrix operand order repaired and checked by reading, not compiled, not run on hardware
//
#include <hip/hip_runtime.h>
#include <math.h>

// ---------------------------------------------------------------------------
// Problem constants
// ---------------------------------------------------------------------------
#define BB    16
#define CIN   80
#define TIN   2048
#define CC    768
#define TP    1024            // T' after stride-2 conv
#define NTOK  (BB * TP)       // 16384 tokens
#define DD    64
#define MM    512
#define HH    256
#define G4H   (4 * HH)        // 1024
#define NMLP  4
#define LN_EPS 1e-5f
#define COMMIT 0.25f

typedef _Float16 half_t;
typedef __attribute__((ext_vector_type(16))) _Float16 v16h;
typedef __attribute__((ext_vector_type(8)))  _Float16 v8h;
typedef __attribute__((ext_vector_type(8)))  float    v8f;

// ---------------------------------------------------------------------------
// WMMA fragment helpers (wave32; A/B use contiguous k order, C/D layout per CDNA5 ISA §7.12.2)
// ---------------------------------------------------------------------------
__device__ __forceinline__ v16h make_frag(v8h lo, v8h hi) {
  v16h f;
#pragma unroll
  for (int i = 0; i < 8; ++i) { f[i] = lo[i]; f[i + 8] = hi[i]; }
  return f;
}

// A-matrix 16x32 f16: lane (l&15) holds row M; halves of lane select K blocks:
// lanes 0-15: K = k0+{0..15}; lanes 16-31: K = k0+{16..31} (contiguous, matches B).
__device__ __forceinline__ v16h load_a_frag(const half_t* A, int lda, int k0) {
  int lane = threadIdx.x & 31;
  const half_t* p = A + (size_t)(lane & 15) * lda + (k0 + ((lane >> 4) << 4));
  return *(const v16h*)p;
}

// B-matrix 32x16 f16 with B[k][n] = W[n][k] (row-major W[Cout][K]):
// lane holds column n = lane&15, K = k0 + 16*(lane>=16) + {0..15} -> contiguous.
__device__ __forceinline__ v16h load_b_frag(const half_t* W, int ldw, int k0) {
  int lane = threadIdx.x & 31;
  return *(const v16h*)(W + (size_t)(lane & 15) * ldw + (k0 + ((lane >> 4) << 4)));
}

// C/D 16x16 f32: lane holds column n = lane&15; VGPR r holds row r + 8*(lane>=16).
__device__ __forceinline__ void store_c(float* Y, int ldc, v8f acc) {
  int lane = threadIdx.x & 31;
  int n = lane & 15;
  int rb = (lane >> 4) << 3;
#pragma unroll
  for (int r = 0; r < 8; ++r) Y[(size_t)(rb + r) * ldc + n] = acc[r];
}

__device__ __forceinline__ v8f wmma_f16(v16h a, v16h b, v8f c) {
  return __builtin_amdgcn_wmma_f32_16x16x32_f16(false, a, false, b, (short)0, c,
                                                false, false);
}

// ---------------------------------------------------------------------------
// Conv1d(CIN->CC, k=4, stride=2, pad=1, no bias) + transpose to [N, C]
// ---------------------------------------------------------------------------
__global__ __launch_bounds__(256) void conv1d_kernel(
    const float* __restrict__ mels, const float* __restrict__ w,
    float* __restrict__ out) {
  int idx = blockIdx.x * 256 + threadIdx.x;
  if (idx >= NTOK * CC) return;
  int c = idx % CC;
  int n = idx / CC;
  int b = n / TP, t = n % TP;
  int p0 = 2 * t - 1;
  float acc = 0.f;
  for (int i = 0; i < CIN; ++i) {
    const float* mrow = mels + ((size_t)b * CIN + i) * TIN;
    const float* wrow = w + ((size_t)c * CIN + i) * 4;
#pragma unroll
    for (int k = 0; k < 4; ++k) {
      int p = p0 + k;
      if (p >= 0 && p < TIN) acc += mrow[p] * wrow[k];
    }
  }
  out[(size_t)n * CC + c] = acc;
}

// ---------------------------------------------------------------------------
// LayerNorm(C=768) + ReLU, fp32 -> f16 ; one block (256 thr) per token row
// ---------------------------------------------------------------------------
__global__ __launch_bounds__(256) void ln_relu_f16_kernel(
    const float* __restrict__ X, const float* __restrict__ gamma,
    const float* __restrict__ beta, half_t* __restrict__ Y) {
  __shared__ float red[256];
  int row = blockIdx.x, tid = threadIdx.x;
  const float* x = X + (size_t)row * CC;
  float v0 = x[tid], v1 = x[tid + 256], v2 = x[tid + 512];
  red[tid] = v0 + v1 + v2;
  __syncthreads();
  for (int o = 128; o > 0; o >>= 1) {
    if (tid < o) red[tid] += red[tid + o];
    __syncthreads();
  }
  float mu = red[0] * (1.f / CC);
  __syncthreads();
  float d0 = v0 - mu, d1 = v1 - mu, d2 = v2 - mu;
  red[tid] = d0 * d0 + d1 * d1 + d2 * d2;
  __syncthreads();
  for (int o = 128; o > 0; o >>= 1) {
    if (tid < o) red[tid] += red[tid + o];
    __syncthreads();
  }
  float rs = rsqrtf(red[0] * (1.f / CC) + LN_EPS);
  half_t* y = Y + (size_t)row * CC;
  y[tid]       = (half_t)fmaxf(d0 * rs * gamma[tid]       + beta[tid],       0.f);
  y[tid + 256] = (half_t)fmaxf(d1 * rs * gamma[tid + 256] + beta[tid + 256], 0.f);
  y[tid + 512] = (half_t)fmaxf(d2 * rs * gamma[tid + 512] + beta[tid + 512], 0.f);
}

// ---------------------------------------------------------------------------
// f32 -> f16 weight conversion
// ---------------------------------------------------------------------------
__global__ __launch_bounds__(256) void cvt_f16_kernel(const float* __restrict__ s,
                                                      half_t* __restrict__ d, int n) {
  int i = blockIdx.x * 256 + threadIdx.x;
  if (i < n) d[i] = (half_t)s[i];
}

// ---------------------------------------------------------------------------
// GEMM: Y[N,Cout] = X[N,K] * W[Cout,K]^T ; f16 inputs, f32 accumulate (WMMA).
// Register-blocked: each wave computes a 32x64 patch (2 row-tiles x 4
// col-tiles = 8 accumulators). Per K-chunk: 2 A-frags + 4 B-frags feed
// 8 WMMAs -> 1.5 b128 loads per WMMA instead of 4. Requires N%32==0,
// Cout%64==0 (768 and 64 both qualify).
// ---------------------------------------------------------------------------
__global__ __launch_bounds__(256) void gemm_wmma_kernel(
    const half_t* __restrict__ X, const half_t* __restrict__ W,
    float* __restrict__ Y, int N, int K, int Cout) {
  int waveId = threadIdx.x >> 5;
  int wid = blockIdx.x * 8 + waveId;
  int colGroups = Cout >> 6;                // 64-wide column groups
  int groups = (N >> 5) * colGroups;        // 32-tall row groups
  if (wid >= groups) return;                // wave-uniform exit
  int row = (wid / colGroups) << 5;
  int col = (wid % colGroups) << 6;
  const half_t* A0 = X + (size_t)row * K;
  const half_t* A1 = A0 + (size_t)16 * K;
  v8f acc[2][4];
#pragma unroll
  for (int rr = 0; rr < 2; ++rr)
#pragma unroll
    for (int cc = 0; cc < 4; ++cc)
      acc[rr][cc] = (v8f){0.f, 0.f, 0.f, 0.f, 0.f, 0.f, 0.f, 0.f};

  for (int k0 = 0; k0 < K; k0 += 32) {
    v16h a0 = load_a_frag(A0, K, k0);
    v16h a1 = load_a_frag(A1, K, k0);
#pragma unroll
    for (int cc = 0; cc < 4; ++cc) {
      v16h b = load_b_frag(W + (size_t)(col + 16 * cc) * K, K, k0);
      acc[0][cc] = wmma_f16(a0, b, acc[0][cc]);
      acc[1][cc] = wmma_f16(a1, b, acc[1][cc]);
    }
  }
#pragma unroll
  for (int cc = 0; cc < 4; ++cc) {
    store_c(Y + (size_t)row * Cout + col + 16 * cc, Cout, acc[0][cc]);
    store_c(Y + (size_t)(row + 16) * Cout + col + 16 * cc, Cout, acc[1][cc]);
  }
}

// ---------------------------------------------------------------------------
// VQ: nearest codebook entry per token; emits quantized fp32 + f16, counts, SSE
// ---------------------------------------------------------------------------
__global__ __launch_bounds__(512) void vq_init_kernel(float* counts, float* sse) {
  int tid = threadIdx.x;
  counts[tid] = 0.f;
  if (tid == 0) *sse = 0.f;
}

__global__ __launch_bounds__(256) void vq_kernel(
    const float* __restrict__ z, const float* __restrict__ emb,
    float* __restrict__ q_out, half_t* __restrict__ q16,
    float* __restrict__ counts, float* __restrict__ sse) {
  int n = blockIdx.x * 256 + threadIdx.x;
  if (n >= NTOK) return;
  float x[DD];
  const float* zr = z + (size_t)n * DD;
#pragma unroll
  for (int d = 0; d < DD; ++d) x[d] = zr[d];
  float best = 3.4e38f;
  int bi = 0;
  for (int m = 0; m < MM; ++m) {
    const float* e = emb + (size_t)m * DD;
    float s = 0.f;
#pragma unroll
    for (int d = 0; d < DD; ++d) {
      float df = x[d] - e[d];
      s += df * df;
    }
    if (s < best) { best = s; bi = m; }     // first-min tie-break like argmin
  }
  const float* e = emb + (size_t)bi * DD;
#pragma unroll
  for (int d = 0; d < DD; ++d) {
    float v = e[d];
    q_out[(size_t)n * DD + d] = v;          // quantized_st == quantized (fwd)
    q16[(size_t)n * DD + d] = (half_t)v;
  }
  atomicAdd(&counts[bi], 1.f);
  atomicAdd(sse, best);                     // best == sum((x-q)^2)
}

__global__ __launch_bounds__(512) void vq_finalize_kernel(
    const float* __restrict__ counts, const float* __restrict__ sse,
    float* __restrict__ loss, float* __restrict__ ppl) {
  __shared__ float red[512];
  int tid = threadIdx.x;
  float a = counts[tid] * (1.f / (float)NTOK);
  red[tid] = a * logf(a + 1e-10f);
  __syncthreads();
  for (int o = 256; o > 0; o >>= 1) {
    if (tid < o) red[tid] += red[tid + o];
    __syncthreads();
  }
  if (tid == 0) {
    *ppl = expf(-red[0]);
    *loss = COMMIT * (*sse) * (1.f / ((float)NTOK * (float)DD));
  }
}

// ---------------------------------------------------------------------------
// LSTM: all 16 batch rows = WMMA M dim; one workgroup of 16 waves.
// Wave w owns hidden columns [16w, 16w+16) for all 4 gates (i,f,g,o), so
// gate mixing stays in registers; h in LDS (A-matrix source), c in registers.
// gates = Q_t * W_ih^T + H * W_hh^T + (b_ih + b_hh), torch gate order i,f,g,o.
// ---------------------------------------------------------------------------
__device__ __forceinline__ float sigm(float x) { return 1.f / (1.f + expf(-x)); }

__global__ __launch_bounds__(512) void lstm_wmma_kernel(
    const half_t* __restrict__ q16,          // [NTOK, DD] f16
    const half_t* __restrict__ wih16,        // [4H, DD]  f16
    const half_t* __restrict__ whh16,        // [4H, HH]  f16
    const float* __restrict__ b_ih, const float* __restrict__ b_hh,
    float* __restrict__ out_h) {             // [B, TP, HH] fp32
  __shared__ __align__(32) half_t hs[BB * HH];   // h state, f16, row-major [b][j]
  int tid = threadIdx.x;
  int lane = tid & 31;
  int w = tid >> 5;                          // wave 0..15
  for (int i = tid; i < BB * HH; i += 512) hs[i] = (half_t)0.f;
  __syncthreads();

  int j = (w << 4) + (lane & 15);            // hidden column this lane owns
  int brow = (lane >> 4) << 3;               // D-matrix row base (0 or 8)
  float cst[8];
#pragma unroll
  for (int r = 0; r < 8; ++r) cst[r] = 0.f;
  float bias[4];
#pragma unroll
  for (int g = 0; g < 4; ++g) bias[g] = b_ih[g * HH + j] + b_hh[g * HH + j];

  for (int t = 0; t < TP; ++t) {
    // A-frags for H (from LDS) : K = 256 -> 8 chunks
    v16h ah[8];
#pragma unroll
    for (int kk = 0; kk < 8; ++kk) ah[kk] = load_a_frag(hs, HH, kk * 32);
    // A-frags for Q_t (global) : rows n = b*TP + t, K = 64 -> 2 chunks
    v16h aq[2];
#pragma unroll
    for (int kk = 0; kk < 2; ++kk) {
      const half_t* p = q16 + ((size_t)(lane & 15) * TP + t) * DD +
                        (kk * 32 + ((lane >> 4) << 4));
      aq[kk] = *(const v16h*)p;
    }
    v8f acc[4];
#pragma unroll
    for (int g = 0; g < 4; ++g) {
      v8f a = {0.f, 0.f, 0.f, 0.f, 0.f, 0.f, 0.f, 0.f};
      const half_t* Bih = wih16 + (size_t)(g * HH + (w << 4)) * DD;
#pragma unroll
      for (int kk = 0; kk < 2; ++kk) a = wmma_f16(aq[kk], load_b_frag(Bih, DD, kk * 32), a);
      const half_t* Bhh = whh16 + (size_t)(g * HH + (w << 4)) * HH;
#pragma unroll
      for (int kk = 0; kk < 8; ++kk) a = wmma_f16(ah[kk], load_b_frag(Bhh, HH, kk * 32), a);
      acc[g] = a;
    }
    __syncthreads();                         // all reads of hs done
#pragma unroll
    for (int r = 0; r < 8; ++r) {
      int b = brow + r;
      float gi = acc[0][r] + bias[0];
      float gf = acc[1][r] + bias[1];
      float gg = acc[2][r] + bias[2];
      float go = acc[3][r] + bias[3];
      float c = sigm(gf) * cst[r] + sigm(gi) * tanhf(gg);
      float h = sigm(go) * tanhf(c);
      cst[r] = c;
      hs[b * HH + j] = (half_t)h;
      out_h[((size_t)b * TP + t) * HH + j] = h;
    }
    __syncthreads();                         // writes visible before next step
  }
}

// ---------------------------------------------------------------------------
// Launch: conv -> 4x(LN+GEMM) -> LN+GEMM(z) -> VQ -> LSTM
// ---------------------------------------------------------------------------
extern "C" void kernel_launch(void* const* d_in, const int* in_sizes, int n_in,
                              void* d_out, int out_size, void* d_ws, size_t ws_size,
                              hipStream_t stream) {
  const float* mels     = (const float*)d_in[0];
  const float* conv_w   = (const float*)d_in[1];
  const float* ln_gamma = (const float*)d_in[2];
  const float* ln_beta  = (const float*)d_in[3];
  const float* w_hidden = (const float*)d_in[4];
  const float* w_out    = (const float*)d_in[5];
  const float* embed    = (const float*)d_in[6];
  const float* w_ih     = (const float*)d_in[7];
  const float* w_hh     = (const float*)d_in[8];
  const float* b_ih     = (const float*)d_in[9];
  const float* b_hh     = (const float*)d_in[10];

  float* out   = (float*)d_out;
  float* out_q = out;                                    // [16,1024,64]
  float* out_c = out + (size_t)NTOK * DD;                // [16,1024,256]
  float* out_z = out + (size_t)NTOK * DD + (size_t)NTOK * HH;
  float* out_loss = out + (size_t)NTOK * DD * 2 + (size_t)NTOK * HH;
  float* out_ppl  = out_loss + 1;

  char* ws = (char*)d_ws;
  size_t off = 0;
  float*  h0    = (float*)(ws + off);  off += (size_t)NTOK * CC * 4;   // 48 MB
  half_t* x16   = (half_t*)(ws + off); off += (size_t)NTOK * CC * 2;   // 24 MB
  half_t* wbuf  = (half_t*)(ws + off); off += (size_t)CC * CC * 2;
  half_t* q16   = (half_t*)(ws + off); off += (size_t)NTOK * DD * 2;
  half_t* wih16 = (half_t*)(ws + off); off += (size_t)G4H * DD * 2;
  half_t* whh16 = (half_t*)(ws + off); off += (size_t)G4H * HH * 2;
  float*  counts= (float*)(ws + off);  off += MM * 4;
  float*  sse   = (float*)(ws + off);  off += 256;

  // 1) Conv1d -> h0 [N, C]
  conv1d_kernel<<<(NTOK * CC + 255) / 256, 256, 0, stream>>>(mels, conv_w, h0);

  // 2) 4 x (LN+ReLU -> f16 ; GEMM 768x768 via WMMA -> h0)
  for (int i = 0; i < NMLP; ++i) {
    ln_relu_f16_kernel<<<NTOK, 256, 0, stream>>>(h0, ln_gamma + i * CC,
                                                 ln_beta + i * CC, x16);
    cvt_f16_kernel<<<(CC * CC + 255) / 256, 256, 0, stream>>>(
        w_hidden + (size_t)i * CC * CC, wbuf, CC * CC);
    gemm_wmma_kernel<<<(NTOK / 32) * (CC / 64) / 8, 256, 0, stream>>>(
        x16, wbuf, h0, NTOK, CC, CC);
  }

  // 3) final LN+ReLU ; GEMM 768->64 -> z_beforeVQ (written straight to d_out)
  ln_relu_f16_kernel<<<NTOK, 256, 0, stream>>>(h0, ln_gamma + NMLP * CC,
                                               ln_beta + NMLP * CC, x16);
  cvt_f16_kernel<<<(DD * CC + 255) / 256, 256, 0, stream>>>(w_out, wbuf, DD * CC);
  gemm_wmma_kernel<<<(NTOK / 32) * (DD / 64) / 8, 256, 0, stream>>>(
      x16, wbuf, out_z, NTOK, CC, DD);

  // 4) VQ: argmin, quantize, loss/perplexity stats
  vq_init_kernel<<<1, 512, 0, stream>>>(counts, sse);
  vq_kernel<<<NTOK / 256, 256, 0, stream>>>(out_z, embed, out_q, q16, counts, sse);
  vq_finalize_kernel<<<1, 512, 0, stream>>>(counts, sse, out_loss, out_ppl);

  // 5) LSTM over quantized (single 16-wave workgroup, WMMA recurrence)
  cvt_f16_kernel<<<(G4H * DD + 255) / 256, 256, 0, stream>>>(w_ih, wih16, G4H * DD);
  cvt_f16_kernel<<<(G4H * HH + 255) / 256, 256, 0, stream>>>(w_hh, whh16, G4H * HH);
  lstm_wmma_kernel<<<1, 512, 0, stream>>>(q16, wih16, whh16, b_ih, b_hh, out_c);
}
